// TernaryLinear_44719199486025
// MI455X (gfx1250) — matrix operand order repaired and checked by reading, not compiled, not run on hardware
//
#include <hip/hip_runtime.h>

typedef __attribute__((ext_vector_type(4)))  float    v4f;
typedef __attribute__((ext_vector_type(8)))  float    v8f;
typedef __attribute__((ext_vector_type(4)))  _Float16 v4h;
typedef __attribute__((ext_vector_type(8)))  _Float16 v8h;
typedef __attribute__((ext_vector_type(16))) _Float16 v16h;
typedef __attribute__((ext_vector_type(4)))  unsigned int u32x4;
typedef __attribute__((ext_vector_type(8)))  int          i32x8;
typedef __attribute__((ext_vector_type(4)))  int          i32x4;

#define D_IN   1024
#define D_OUT  1024
#define M_TOT  32768

#define MT   128   // M tile per block
#define NT   128   // N tile per block
#define KT   32    // K step
#define LDH  40    // padded LDS row stride in halves (32 data + 8 pad = 80 B rows)

// Tensor Data Mover path only in device compilation and only if builtin exists.
#if defined(__HIP_DEVICE_COMPILE__) && __has_builtin(__builtin_amdgcn_tensor_load_to_lds)
#define HAVE_TDM 1
#else
#define HAVE_TDM 0
#endif

// ---------------- kernel 0: zero the reduction slot ----------------
__global__ void k_init(float* sum) {
    if (threadIdx.x == 0) *sum = 0.0f;
}

// ---------------- kernel 1: sum(|W|) ----------------
__global__ __launch_bounds__(256) void k_abssum(const float* __restrict__ w,
                                                float* __restrict__ sum) {
    __shared__ float red[256];
    const int t = threadIdx.x;
    const v4f x = ((const v4f*)w)[(size_t)blockIdx.x * 256 + t];
    red[t] = fabsf(x[0]) + fabsf(x[1]) + fabsf(x[2]) + fabsf(x[3]);
    __syncthreads();
    for (int off = 128; off > 0; off >>= 1) {
        if (t < off) red[t] += red[t + off];
        __syncthreads();
    }
    if (t == 0) atomicAdd(sum, red[0]);
}

// ---------------- kernel 2: ternary quantize -> f16 * alpha ----------------
__global__ __launch_bounds__(256) void k_quant(const float* __restrict__ w,
                                               const float* __restrict__ alpha,
                                               const float* __restrict__ sum,
                                               _Float16* __restrict__ wq) {
    const size_t i = (size_t)blockIdx.x * 256 + threadIdx.x;  // float4 index
    const float delta = 0.7f * (*sum) * (1.0f / (float)(D_IN * D_OUT));
    const float a = *alpha;
    const v4f x = ((const v4f*)w)[i];
    v4h q;
#pragma unroll
    for (int j = 0; j < 4; ++j) {
        const float v = x[j];
        const float r = (v > delta) ? a : ((v < -delta) ? -a : 0.0f);
        q[j] = (_Float16)r;
    }
    ((v4h*)wq)[i] = q;
}

#if HAVE_TDM
// TDM: load one 128-row x 32-half B tile (row stride D_IN halves) into LDS with
// per-row padding of 4 DWORDs after every 16 DWORDs -> 80-byte LDS rows (== LDH).
__device__ __forceinline__ void tdm_load_B(const _Float16* gsrc, const _Float16* ldst) {
    const unsigned long long ga = (unsigned long long)(size_t)gsrc;
    const unsigned lds = (unsigned)(size_t)ldst;   // low 32 bits = LDS byte offset
    u32x4 g0;
    g0[0] = 1u;                                            // count=1, user descriptor
    g0[1] = lds;                                           // lds_addr
    g0[2] = (unsigned)ga;                                  // global_addr[31:0]
    g0[3] = (unsigned)((ga >> 32) & 0x01FFFFFFu) | (2u << 30); // global_addr[56:32] | type=2
    i32x8 g1;
    g1[0] = (int)((1u << 16)      // data_size = 2 bytes
                | (1u << 20)      // pad_enable
                | (3u << 22)      // pad_interval = 16 DWORDs (one 64-byte row)
                | (3u << 25));    // pad_amount  = 4 DWORDs (16 bytes)
    g1[1] = 0;                    // abar addr / tensor_dim0 low16 (dim0 = 1<<20)
    g1[2] = 0x10;                 // tensor_dim0 hi16 = 16 -> tensor_dim0 = 1<<20
    g1[3] = 0x10 | (32 << 16);    // tensor_dim1 = 1<<20, tile_dim0 = 32
    g1[4] = 128;                  // tile_dim1 = 128, tile_dim2 = 0
    g1[5] = D_IN;                 // tensor_dim0_stride = 1024 elements
    g1[6] = 0;
    g1[7] = 0;
    i32x4 z4 = {0, 0, 0, 0};
#if defined(__clang_major__) && (__clang_major__ >= 23)
    i32x8 z8 = {0, 0, 0, 0, 0, 0, 0, 0};
    __builtin_amdgcn_tensor_load_to_lds(g0, g1, z4, z4, z8, 0);
#else
    __builtin_amdgcn_tensor_load_to_lds(g0, g1, z4, z4, 0);
#endif
}
#endif

// ---------------- kernel 3: WMMA GEMM  out = A(f32->f16) * Wq^T + bias ----------------
// Ping-pong LDS double buffering: B tiles arrive via the Tensor Data Mover
// (one tensor_load_to_lds per block per K-step, issued by wave 0, TENSORcnt),
// A tiles are prefetched to VGPRs (overlapped with WMMA), converted f32->f16,
// then stored to the back buffer. One workgroup barrier per K-step.
__global__ __launch_bounds__(256) void k_gemm(const float* __restrict__ A,
                                              const _Float16* __restrict__ Wq,
                                              const float* __restrict__ bias,
                                              float* __restrict__ out) {
    __shared__ _Float16 lA[2][MT * LDH];
    __shared__ _Float16 lB[2][NT * LDH];

    const int tid  = threadIdx.x;
    const int lane = tid & 31;
    const int wv   = tid >> 5;        // 0..7
    const int wM   = wv >> 2;         // 0..1
    const int wN   = wv & 3;          // 0..3
    const int l16  = lane & 15;
    const int hsel = lane >> 4;       // 0/1

    const int tileN = blockIdx.x * NT;
    const int tileM = blockIdx.y * MT;

    const int sRow = tid >> 1;        // 0..127
    const int sH   = tid & 1;         // 16-element half of the K-step

    const float*    gA  = A  + (size_t)(tileM + sRow) * D_IN + sH * 16;
    const _Float16* gB0 = Wq + (size_t)tileN * D_IN;               // TDM tile origin
#if !HAVE_TDM
    const _Float16* gBr = Wq + (size_t)(tileN + sRow) * D_IN + sH * 16;
#endif

    v8f acc[4][2] = {};

    // ---------------- prologue: stage tile 0 into buffer 0 ----------------
    {
        v4f a0 = *(const v4f*)(gA + 0);
        v4f a1 = *(const v4f*)(gA + 4);
        v4f a2 = *(const v4f*)(gA + 8);
        v4f a3 = *(const v4f*)(gA + 12);
#if HAVE_TDM
        if (tid < 32) tdm_load_B(gB0, &lB[0][0]);
#else
        v8h b0 = *(const v8h*)(gBr + 0);
        v8h b1 = *(const v8h*)(gBr + 8);
        *(v8h*)&lB[0][sRow * LDH + sH * 16 + 0] = b0;
        *(v8h*)&lB[0][sRow * LDH + sH * 16 + 8] = b1;
#endif
        v8h ha, hb;
#pragma unroll
        for (int j = 0; j < 4; ++j) { ha[j] = (_Float16)a0[j]; ha[4 + j] = (_Float16)a1[j]; }
#pragma unroll
        for (int j = 0; j < 4; ++j) { hb[j] = (_Float16)a2[j]; hb[4 + j] = (_Float16)a3[j]; }
        *(v8h*)&lA[0][sRow * LDH + sH * 16 + 0] = ha;
        *(v8h*)&lA[0][sRow * LDH + sH * 16 + 8] = hb;
#if HAVE_TDM
        if (tid < 32) __builtin_amdgcn_s_wait_tensorcnt(0);
#endif
        __syncthreads();
    }

    int p = 0;
    for (int k0 = 0; k0 < D_IN; k0 += KT) {
        const bool nxt = (k0 + KT) < D_IN;

        // ---- prefetch next tile (global -> regs / TDM -> back LDS buffer) ----
        v4f a0, a1, a2, a3;
        if (nxt) {
            a0 = *(const v4f*)(gA + k0 + KT + 0);
            a1 = *(const v4f*)(gA + k0 + KT + 4);
            a2 = *(const v4f*)(gA + k0 + KT + 8);
            a3 = *(const v4f*)(gA + k0 + KT + 12);
#if HAVE_TDM
            if (tid < 32) tdm_load_B(gB0 + (k0 + KT), &lB[p ^ 1][0]);
#endif
        }
#if !HAVE_TDM
        v8h b0, b1;
        if (nxt) {
            b0 = *(const v8h*)(gBr + k0 + KT + 0);
            b1 = *(const v8h*)(gBr + k0 + KT + 8);
        }
#endif

        // ---- compute from front buffer p ----
        v16h bf[2];
#pragma unroll
        for (int nt = 0; nt < 2; ++nt) {
            const _Float16* q = &lB[p][(wN * 32 + nt * 16 + l16) * LDH + hsel * 16];
            v8h x0 = *(const v8h*)(q);
            v8h x1 = *(const v8h*)(q + 8);
            bf[nt] = __builtin_shufflevector(x0, x1,
                     0, 1, 2, 3, 4, 5, 6, 7, 8, 9, 10, 11, 12, 13, 14, 15);
        }
#pragma unroll
        for (int mt = 0; mt < 4; ++mt) {
            const _Float16* q = &lA[p][(wM * 64 + mt * 16 + l16) * LDH];
            v8h x0 = *(const v8h*)(q + hsel * 16);
            v8h x1 = *(const v8h*)(q + hsel * 16 + 8);
            v16h af = __builtin_shufflevector(x0, x1,
                      0, 1, 2, 3, 4, 5, 6, 7, 8, 9, 10, 11, 12, 13, 14, 15);
#pragma unroll
            for (int nt = 0; nt < 2; ++nt) {
                acc[mt][nt] = __builtin_amdgcn_wmma_f32_16x16x32_f16(
                    false, af, false, bf[nt], (short)0, acc[mt][nt], false, false);
            }
        }

        // ---- drain prefetch into back buffer ----
        if (nxt) {
            v8h ha, hb;
#pragma unroll
            for (int j = 0; j < 4; ++j) { ha[j] = (_Float16)a0[j]; ha[4 + j] = (_Float16)a1[j]; }
#pragma unroll
            for (int j = 0; j < 4; ++j) { hb[j] = (_Float16)a2[j]; hb[4 + j] = (_Float16)a3[j]; }
            *(v8h*)&lA[p ^ 1][sRow * LDH + sH * 16 + 0] = ha;
            *(v8h*)&lA[p ^ 1][sRow * LDH + sH * 16 + 8] = hb;
#if HAVE_TDM
            if (tid < 32) __builtin_amdgcn_s_wait_tensorcnt(0);
#else
            *(v8h*)&lB[p ^ 1][sRow * LDH + sH * 16 + 0] = b0;
            *(v8h*)&lB[p ^ 1][sRow * LDH + sH * 16 + 8] = b1;
#endif
        }
        __syncthreads();
        p ^= 1;
    }

    // ---------------- epilogue: add bias, store fp32 ----------------
#pragma unroll
    for (int nt = 0; nt < 2; ++nt) {
        const int col = tileN + wN * 32 + nt * 16 + l16;
        const float bv = bias[col];
#pragma unroll
        for (int mt = 0; mt < 4; ++mt) {
            const int rbase = tileM + wM * 64 + mt * 16 + hsel * 8;
#pragma unroll
            for (int r = 0; r < 8; ++r) {
                out[(size_t)(rbase + r) * D_OUT + col] = acc[mt][nt][r] + bv;
            }
        }
    }
}

extern "C" void kernel_launch(void* const* d_in, const int* in_sizes, int n_in,
                              void* d_out, int out_size, void* d_ws, size_t ws_size,
                              hipStream_t stream) {
    const float* input  = (const float*)d_in[0];   // [8, 4096, 1024] -> [32768, 1024]
    const float* weight = (const float*)d_in[1];   // [1024, 1024]
    const float* alpha  = (const float*)d_in[2];   // [1]
    const float* bias   = (const float*)d_in[3];   // [1024]
    float* out = (float*)d_out;

    float*    ws_sum = (float*)d_ws;
    _Float16* wq     = (_Float16*)((char*)d_ws + 256);   // 2 MB f16 quantized weights

    (void)in_sizes; (void)n_in; (void)out_size; (void)ws_size;

    k_init<<<1, 64, 0, stream>>>(ws_sum);
    k_abssum<<<(D_IN * D_OUT) / (256 * 4), 256, 0, stream>>>(weight, ws_sum);
    k_quant<<<(D_IN * D_OUT) / (256 * 4), 256, 0, stream>>>(weight, alpha, ws_sum, wq);

    dim3 grid(D_OUT / NT, M_TOT / MT);   // (8, 256)
    k_gemm<<<grid, 256, 0, stream>>>(input, wq, bias, out);
}
